// MFVIConstituency_33904471834777
// MI455X (gfx1250) — matrix operand order repaired and checked by reading, not compiled, not run on hardware
//
#include <hip/hip_runtime.h>
#include <hip/hip_bf16.h>
#include <stdint.h>

// MFVI constituency: q_{t+1}[b,i,j] = s_span + sum_k sigmoid(q_t)[b,i,k] * sp[b,i,j,k]
// sp = s_pair * mask2o. Batched GEMV over B*S independent 128x128 matrices, mapped
// onto v_wmma_f32_16x16x32_f16 with the q-vector replicated across all 16 B-columns.
//
// Masking strategy (branchless inner loop):
//   - span[b,j,k] = mask|mask^T stored as f16 {1,0}: pure multiply in the k-loop.
//   - k==min(i,j) / k==max(i,j) exclusions (<=2 of 128 k's): subtracted at writeback.
//   - mask[b,i,j] (row constant): applied as a scale on acc at writeback.

typedef __attribute__((ext_vector_type(16))) _Float16 v16h;
typedef __attribute__((ext_vector_type(8)))  _Float16 v8h;
typedef __attribute__((ext_vector_type(8)))  float    v8f;
typedef __attribute__((ext_vector_type(4)))  float    v4f;

#define B_DIM 16
#define S_DIM 128

__device__ __forceinline__ float sigmoidf(float x) {
    return 1.0f / (1.0f + __expf(-x));
}

// Pass 1: spanh[b,j,k] = (mask[b,j,k] | mask[b,k,j]) ? 1.0h : 0.0h   (512 KB)
//         qsig0[b,i,k] = (f16) sigmoid(s_span[b,i,k])                (512 KB)
__global__ void mfvi_prep_kernel(const float* __restrict__ s_span,
                                 const uint8_t* __restrict__ mask,
                                 _Float16* __restrict__ spanh,
                                 _Float16* __restrict__ qsig) {
    int idx = blockIdx.x * blockDim.x + threadIdx.x;   // over B*S*S
    if (idx >= B_DIM * S_DIM * S_DIM) return;
    int k = idx & (S_DIM - 1);
    int j = (idx >> 7) & (S_DIM - 1);
    int b = idx >> 14;
    uint8_t m1 = mask[idx];                                   // mask[b,j,k]
    uint8_t m2 = mask[((size_t)(b * S_DIM + k)) * S_DIM + j]; // mask[b,k,j]
    spanh[idx] = (m1 | m2) ? (_Float16)1.0f : (_Float16)0.0f;
    qsig[idx]  = (_Float16)sigmoidf(s_span[idx]);
}

// One block per (b,i); wave w handles j-tile [16w, 16w+16).
// acc(16x16 f32) += A(16j x 32k : (f16)s_pair * span, per ISA operand layout)
//                 x B(32k x 16 : sigmoid(q) replicated in all columns).
__global__ __launch_bounds__(256)
void mfvi_iter_kernel(const float* __restrict__ s_pair,
                      const float* __restrict__ s_span,
                      const uint8_t* __restrict__ mask,
                      const _Float16* __restrict__ spanh,
                      const _Float16* __restrict__ qin,
                      _Float16* __restrict__ qout,
                      float* __restrict__ finalOut) {
    const int bi   = blockIdx.x;              // b*S + i
    const int i    = bi & (S_DIM - 1);
    const int b    = bi >> 7;
    const int wave = threadIdx.x >> 5;
    const int lane = threadIdx.x & 31;
    const int j0   = wave * 16;
    const int jrow = j0 + (lane & 15);        // A-operand row this lane feeds
    const bool hiHalf = lane >= 16;
    const int off0 = hiHalf ? 8 : 0;          // A-operand K striping (ISA 16-bit 16x32)

    const float*    sp      = s_pair + (((size_t)bi * S_DIM + jrow) * S_DIM);
    const _Float16* spanRow = spanh + ((size_t)(b * S_DIM + jrow)) * S_DIM;
    const _Float16* qrow    = qin + (size_t)bi * S_DIM;      // sigmoid(q)[b,i,:]

    v8f acc = {};
#pragma unroll
    for (int k0 = 0; k0 < S_DIM; k0 += 32) {
        // A halves 0..7 <- k = k0+off0+0..7 ; halves 8..15 <- k0+off0+16..23
        v4f a0 = *(const v4f*)(sp + k0 + off0);
        v4f a1 = *(const v4f*)(sp + k0 + off0 + 4);
        v4f a2 = *(const v4f*)(sp + k0 + off0 + 16);
        v4f a3 = *(const v4f*)(sp + k0 + off0 + 20);
        v8h s0 = *(const v8h*)(spanRow + k0 + off0);
        v8h s1 = *(const v8h*)(spanRow + k0 + off0 + 16);
        v16h A;
#pragma unroll
        for (int t = 0; t < 8; ++t) {
            float v = (t < 4) ? a0[t] : a1[t - 4];
            A[t] = (_Float16)v * s0[t];
        }
#pragma unroll
        for (int t = 0; t < 8; ++t) {
            float v = (t < 4) ? a2[t] : a3[t - 4];
            A[8 + t] = (_Float16)v * s1[t];
        }
        // B (all 16 columns identical): same K striping as A: k0+off0+0..7, k0+off0+16..23
        v8h q0 = *(const v8h*)(qrow + k0 + off0);
        v8h q1 = *(const v8h*)(qrow + k0 + off0 + 16);
        v16h Bm = __builtin_shufflevector(q0, q1, 0, 1, 2, 3, 4, 5, 6, 7, 8, 9, 10, 11, 12, 13, 14, 15);
        acc = __builtin_amdgcn_wmma_f32_16x16x32_f16(
            /*neg_a=*/false, A, /*neg_b=*/false, Bm,
            /*c_mod=*/(short)0, acc, /*reuse_a=*/false, /*reuse_b=*/false);
    }

    // D layout: VGPR r holds M=r (lanes 0-15) / M=r+8 (lanes 16-31); all columns equal.
    // Lanes with (lane&15)<8 each own one output row j.
    const int r = lane & 15;
    if (r < 8) {
        const int j = j0 + (hiHalf ? 8 : 0) + r;
        const size_t o = (size_t)bi * S_DIM + j;          // [b,i,j] flat
        const int lo = min(i, j);
        const int hi = max(i, j);
        const float*    spRowJ   = s_pair + (((size_t)bi * S_DIM + j) * S_DIM);
        const _Float16* spanRowJ = spanh + ((size_t)(b * S_DIM + j)) * S_DIM;
        // Subtract the k==lo / k==hi terms exactly as the WMMA included them
        // (f16 operands -> f16 x f16 product is exact in f32).
        float corr = (float)qrow[lo] * (float)(_Float16)spRowJ[lo] * (float)spanRowJ[lo];
        if (hi != lo)
            corr += (float)qrow[hi] * (float)(_Float16)spRowJ[hi] * (float)spanRowJ[hi];
        const float mij = mask[o] ? 1.0f : 0.0f;           // mask[b,i,j]
        float qn = s_span[o] + mij * (acc[r] - corr);
        float sg = sigmoidf(qn);
        qout[o] = (_Float16)sg;
        if (finalOut) finalOut[o] = sg;
    }
}

extern "C" void kernel_launch(void* const* d_in, const int* in_sizes, int n_in,
                              void* d_out, int out_size, void* d_ws, size_t ws_size,
                              hipStream_t stream) {
    const float*   s_span = (const float*)d_in[0];
    const float*   s_pair = (const float*)d_in[1];
    const uint8_t* mask   = (const uint8_t*)d_in[2];   // jnp.bool_: 1 byte/elem
    float* out = (float*)d_out;

    uint8_t*   ws    = (uint8_t*)d_ws;
    _Float16*  spanh = (_Float16*)ws;                        // 512 KB
    _Float16*  qA    = (_Float16*)(ws + 524288);             // 512 KB
    _Float16*  qB    = (_Float16*)(ws + 524288 + 524288);    // 512 KB

    const int n = B_DIM * S_DIM * S_DIM;
    mfvi_prep_kernel<<<(n + 255) / 256, 256, 0, stream>>>(s_span, mask, spanh, qA);

    dim3 grid(B_DIM * S_DIM);   // one block per (b,i)
    dim3 block(256);            // 8 wave32 = 8 j-tiles of 16
    mfvi_iter_kernel<<<grid, block, 0, stream>>>(s_pair, s_span, mask, spanh, qA, qB, nullptr);
    mfvi_iter_kernel<<<grid, block, 0, stream>>>(s_pair, s_span, mask, spanh, qB, qA, nullptr);
    mfvi_iter_kernel<<<grid, block, 0, stream>>>(s_pair, s_span, mask, spanh, qA, qB, out);
}
